// CompressedSensing_51479478010600
// MI455X (gfx1250) — matrix operand order repaired and checked by reading, not compiled, not run on hardware
//
#include <hip/hip_runtime.h>

typedef __attribute__((ext_vector_type(16))) _Float16     v16h;
typedef __attribute__((ext_vector_type(8)))  _Float16     v8h;
typedef __attribute__((ext_vector_type(8)))  float        v8f;
typedef __attribute__((ext_vector_type(8)))  unsigned int v8u;
typedef __attribute__((ext_vector_type(4)))  unsigned int v4u;

#define N_CODE  5329
#define N_PIX   81
#define KPAD    5376          // N_CODE padded to multiple of 64 (168 k-tiles)
#define MPAD    96            // N_PIX padded to multiple of 32
#define KT1     (KPAD/32)     // 168 K-tiles in step 1
#define KH1     (KT1/2)       // 84  (dual accumulation chains)
#define NT1     (MPAD/16)     // 6   N-tiles in step 1
#define KT2     (MPAD/32)     // 3   K-tiles in step 2
#define NT2     (KPAD/16)     // 336 N-tiles in step 2 (divisible by 16)
#define N_ITERS 100

// padded LDS strides: byte stride multiple of 16 (for ds_load_b128) and the
// dword stride mod 64 has gcd 4 with 64 so 16 rows tile all 64 banks.
#define YSTRIDE 5384          // halves; 10768 B = 673*16, 2692 dw % 64 = 4
#define VSTRIDE 104           // halves; 208 B = 13*16,   52 dw % 64 = 52
#define ISTRIDE 100           // floats; 400 B,           100 dw % 64 = 36

// ---------------------------------------------------------------------------
// Prep: pack mat into WMMA operand order, f16.
//  matB1 = B-operand fragments of (-Mat) for step 1 (V = Im + Y*(-Mat)):
//    matB1[kt][nt][lane][h] = -mat[n][m],
//      n = kt*32 + ((lane&16)?8:0) + h + ((h&8)?8:0) , m = nt*16 + (lane&15)
//  matT2 = A-operand fragments of Mat row-tiles for step 2 (Re^T = Mat*V^T):
//    matT2[nt][kt][lane][h] = mat[n][m],
//      n = nt*16 + (lane&15) , m = kt*32 + ((lane&16)?8:0) + h + ((h&8)?8:0)
// ---------------------------------------------------------------------------
__global__ void fista_prep(const float* __restrict__ mat,
                           _Float16* __restrict__ matB1,
                           _Float16* __restrict__ matT2) {
  int idx = blockIdx.x * blockDim.x + threadIdx.x;
  const int n1 = KT1 * NT1 * 512;
  if (idx < n1) {
    int h    = idx & 15;
    int lane = (idx >> 4) & 31;
    int nt   = (idx >> 9) % NT1;
    int kt   = idx / (512 * NT1);
    int n    = kt * 32 + ((lane & 16) ? 8 : 0) + h + ((h & 8) ? 8 : 0);
    int m    = nt * 16 + (lane & 15);
    float val = (n < N_CODE && m < N_PIX) ? -mat[n * N_PIX + m] : 0.0f;
    matB1[idx] = (_Float16)val;
  } else {
    int idx2 = idx - n1;
    const int n2 = NT2 * KT2 * 512;
    if (idx2 < n2) {
      int h    = idx2 & 15;
      int lane = (idx2 >> 4) & 31;
      int kt   = (idx2 >> 9) % KT2;
      int nt   = idx2 / (512 * KT2);
      int n    = nt * 16 + (lane & 15);
      int m    = kt * 32 + ((lane & 16) ? 8 : 0) + h + ((h & 8) ? 8 : 0);
      float val = (m < N_PIX && n < N_CODE) ? mat[n * N_PIX + m] : 0.0f;
      matT2[idx2] = (_Float16)val;
    }
  }
}

// ---------------------------------------------------------------------------
// FISTA: 96 blocks x 16 pairs; Y resident in LDS (f16) for all 100 iters.
// ---------------------------------------------------------------------------
__global__ __launch_bounds__(256, 1)
void fista_kernel(const float* __restrict__ x,
                  const _Float16* __restrict__ matB1,
                  const _Float16* __restrict__ matT2,
                  const float* __restrict__ lam,
                  const float* __restrict__ mu,
                  const float* __restrict__ tp,
                  float* __restrict__ out) {
  extern __shared__ char smem[];
  _Float16* Yh = (_Float16*)smem;                                         // [16][YSTRIDE]
  _Float16* Vh = (_Float16*)(smem + 16 * YSTRIDE * 2);                    // [16][VSTRIDE]
  float*    Im = (float*)   (smem + 16 * YSTRIDE * 2 + 16 * VSTRIDE * 2); // [16][ISTRIDE]

  const int tid   = threadIdx.x;
  const int lane  = tid & 31;
  const int wave  = tid >> 5;
  const int ptile = blockIdx.x;

  const float muv    = mu[0];
  const float thr    = lam[0] * 0.005f / muv;
  const float inv_mu = 1.0f / muv;
  const float tpar   = tp[0];

  // packed f16 constants for the prox epilogue
  const _Float16 thrH = (_Float16)thr, imuH = (_Float16)inv_mu;
  const v8h thr8  = {thrH, thrH, thrH, thrH, thrH, thrH, thrH, thrH};
  const v8h imu8  = {imuH, imuH, imuH, imuH, imuH, imuH, imuH, imuH};
  const v8h zero8 = {};

  // zero the Y iterate (including padding)
  unsigned int* Yu = (unsigned int*)Yh;
  for (int i = tid; i < 16 * YSTRIDE / 2; i += 256) Yu[i] = 0u;

  // stage Im = transpose(x,(0,3,1,2)).reshape(.,.,81), zero-padded
  for (int i = tid; i < 16 * ISTRIDE; i += 256) {
    int p = i / ISTRIDE, m = i % ISTRIDE;
    float v = 0.0f;
    if (m < N_PIX) {
      int pg = ptile * 16 + p;
      int b = pg / 3, c = pg % 3;
      int hh = m / 9, ww = m % 9;
      v = x[((b * 9 + hh) * 9 + ww) * 3 + c];
    }
    Im[i] = v;
  }
  __syncthreads();

  const int rowA  = lane & 15;              // A/B-operand row/col (lane&15)
  const int hiOfs = (lane & 16) ? 8 : 0;    // A-operand per-lane K base (halves)
  const int rowHi = (lane & 16) ? 8 : 0;    // C/D layout: row offset for hi half-wave
  const int colC  = lane & 15;              // C/D layout: column

  // 16-bit fragment = two contiguous 16B runs per lane, `gap` halves apart
  auto loadFrag = [&](const _Float16* base, int gap) -> v16h {
    v4u lo = *(const v4u*)(base);
    v4u hi = *(const v4u*)(base + gap);
    v8u au;
#pragma unroll
    for (int i = 0; i < 4; ++i) { au[i] = lo[i]; au[i + 4] = hi[i]; }
    return __builtin_bit_cast(v16h, au);
  };

  float tcur = 1.0f;
  for (int iter = 0; iter < N_ITERS; ++iter) {
    const float tn   = 0.5f * (1.0f + sqrtf(1.0f + 4.0f * tcur * tcur));
    const float coef = (tpar - 1.0f) / tn;  // == 0 for given inputs
    tcur = tn;
    const _Float16 coefH = (_Float16)coef;
    const v8h coef8 = {coefH, coefH, coefH, coefH, coefH, coefH, coefH, coefH};

    // ---- Step 1: V[16p x 96m] = Im + Y*(-Mat); waves 0..5, dual wmma chains
    if (wave < NT1) {
      const int mt = wave;
      v8f acc0, acc1 = {};
#pragma unroll
      for (int r = 0; r < 8; ++r)
        acc0[r] = Im[(r + rowHi) * ISTRIDE + mt * 16 + colC];

      const _Float16* ya  = Yh + rowA * YSTRIDE + hiOfs;
      const _Float16* bp0 = matB1 + (size_t)mt * 512 + lane * 16;
      const _Float16* bp1 = bp0 + (size_t)KH1 * NT1 * 512;
      for (int kt = 0; kt < KH1; ++kt) {
        v16h a0 = loadFrag(ya + kt * 32, 16);          // ds_load_b128 x2
        v16h a1 = loadFrag(ya + (kt + KH1) * 32, 16);
        v8u bu0 = *(const v8u*)bp0;                    // global_load_b128 x2
        v8u bu1 = *(const v8u*)bp1;
        acc0 = __builtin_amdgcn_wmma_f32_16x16x32_f16(false, a0, false,
                 __builtin_bit_cast(v16h, bu0), (short)0, acc0, false, false);
        acc1 = __builtin_amdgcn_wmma_f32_16x16x32_f16(false, a1, false,
                 __builtin_bit_cast(v16h, bu1), (short)0, acc1, false, false);
        bp0 += (size_t)NT1 * 512;
        bp1 += (size_t)NT1 * 512;
      }
#pragma unroll
      for (int r = 0; r < 8; ++r)
        Vh[(r + rowHi) * VSTRIDE + mt * 16 + colC] = (_Float16)(acc0[r] + acc1[r]);
    }
    __syncthreads();

    // ---- Step 2 (transposed): Re^T[16n x 16p] = MatTile * V^T, fused prox.
    // B operand (V^T): lane&15 = pair, two 8-half runs of Vh row, 16 halves apart.
    v16h b2[3];  // register-resident across all N-tiles of this wave
#pragma unroll
    for (int kt = 0; kt < KT2; ++kt)
      b2[kt] = loadFrag(Vh + rowA * VSTRIDE + kt * 32 + ((lane & 16) ? 8 : 0), 16);

    const bool last = (iter == N_ITERS - 1);
    auto prox = [&](int nt, const v8f& acc) {
      const int nbase = nt * 16 + rowHi;              // 8 consecutive n per lane
      _Float16* yp = Yh + colC * YSTRIDE + nbase;     // 16B aligned
      v4u yv = *(const v4u*)yp;                       // ds_load_b128
      v8h yold = __builtin_bit_cast(v8h, yv);
      v8h re   = __builtin_convertvector(acc, v8h);
      v8h w    = yold + re * imu8;
      v8h aw   = __builtin_elementwise_abs(w) - thr8;
      aw       = __builtin_elementwise_max(aw, zero8);
      v8h ynew = __builtin_elementwise_copysign(aw, w);    // soft threshold
      v8h ynxt = ynew;
      if (coef != 0.0f) ynxt = ynew + coef8 * (ynew - yold);
      *(v4u*)yp = __builtin_bit_cast(v4u, ynxt);      // ds_store_b128
      if (last) {
        int pg = ptile * 16 + colC;
        int b = pg / 3, c = pg % 3;
#pragma unroll
        for (int r = 0; r < 8; ++r) {
          int n = nbase + r;
          if (n < N_CODE) out[((size_t)b * N_CODE + n) * 3 + c] = (float)ynew[r];
        }
      }
    };

    for (int nt = wave; nt < NT2; nt += 16) {   // two N-tiles per pass: nt, nt+8
      const int ntB = nt + 8;
      v8f acc0 = {}, acc1 = {};
      const _Float16* apA = matT2 + (size_t)nt  * KT2 * 512 + lane * 16;
      const _Float16* apB = matT2 + (size_t)ntB * KT2 * 512 + lane * 16;
#pragma unroll
      for (int kt = 0; kt < KT2; ++kt) {
        v8u auA = *(const v8u*)(apA + (size_t)kt * 512);   // global_load_b128 x2
        v8u auB = *(const v8u*)(apB + (size_t)kt * 512);
        acc0 = __builtin_amdgcn_wmma_f32_16x16x32_f16(false,
                 __builtin_bit_cast(v16h, auA), false, b2[kt], (short)0, acc0, false, false);
        acc1 = __builtin_amdgcn_wmma_f32_16x16x32_f16(false,
                 __builtin_bit_cast(v16h, auB), false, b2[kt], (short)0, acc1, false, false);
      }
      prox(nt, acc0);
      prox(ntB, acc1);
    }
    __syncthreads();
  }
}

extern "C" void kernel_launch(void* const* d_in, const int* in_sizes, int n_in,
                              void* d_out, int out_size, void* d_ws, size_t ws_size,
                              hipStream_t stream) {
  const float* x   = (const float*)d_in[0];   // [512,9,9,3]
  const float* mat = (const float*)d_in[1];   // [5329,81]
  const float* lam = (const float*)d_in[2];
  const float* mu  = (const float*)d_in[3];
  const float* tp  = (const float*)d_in[4];
  float* out = (float*)d_out;                 // [512,5329,3]

  _Float16* matB1 = (_Float16*)d_ws;                      // 1.03 MB
  _Float16* matT2 = matB1 + (size_t)KT1 * NT1 * 512;      // 1.03 MB

  const int total = (KT1 * NT1 + NT2 * KT2) * 512;
  fista_prep<<<(total + 255) / 256, 256, 0, stream>>>(mat, matB1, matT2);

  const size_t smem = (size_t)16 * YSTRIDE * 2   // Y iterate, f16 (~172 KB)
                    + (size_t)16 * VSTRIDE * 2   // V scratch, f16
                    + (size_t)16 * ISTRIDE * 4;  // Im, f32    (~182 KB total)
  fista_kernel<<<96, 256, smem, stream>>>(x, matB1, matT2, lam, mu, tp, out);
}
